// PT_MotionDeformableAttention_34291018891938
// MI455X (gfx1250) — matrix operand order repaired and checked by reading, not compiled, not run on hardware
//
#include <hip/hip_runtime.h>
#include <hip/hip_bf16.h>

// CDNA5 / gfx1250, wave32. Matrix math via v_wmma_f32_16x16x32_f16
// (f32 accumulate, f16 operands converted in registers).
// Wave tile for the big GEMMs: 16x64 (4 accumulators per wave); all four
// B fragments are fetched from LDS before the four WMMAs so the matrix ops
// issue back-to-back under a single s_wait_dscnt.

typedef __attribute__((ext_vector_type(16))) _Float16 v16h;
typedef __attribute__((ext_vector_type(8)))  float    v8f;

namespace cfg {
constexpr int D  = 256, NH = 8, NL = 1, NP = 4, NS = 12, Dh = 32;
constexpr int HB = 200, WB = 200, BS = 2, A = 256, M = 6;
constexpr int NQ   = A * M;                 // 1536
constexpr int V    = HB * WB;               // 40000
constexpr int Q    = NQ * NS;               // 18432
constexpr int ROWS = BS * NQ;               // 3072
constexpr int NOFF = NH * NS * NL * NP * 2; // 768
constexpr int NATT = NH * NS * NL * NP;     // 384
constexpr int KOUT = NS * D;                // 3072
}

__device__ __forceinline__ float4 add4(float4 a, float4 b) {
  return make_float4(a.x + b.x, a.y + b.y, a.z + b.z, a.w + b.w);
}

// Pack 16 fp32 (k = kbase..kbase+15) into the per-lane
// A fragment of V_WMMA_F32_16X16X32_F16 (A is 16x32 f16, row = lane&15,
// kbase = 16*(lane>>4); this code's contiguous k order, B uses the same).
__device__ __forceinline__ v16h cvt_frag(float4 f0, float4 f1, float4 f2, float4 f3) {
  v16h a;
  a[0]=(_Float16)f0.x; a[1]=(_Float16)f0.y; a[2]=(_Float16)f0.z; a[3]=(_Float16)f0.w;
  a[4]=(_Float16)f1.x; a[5]=(_Float16)f1.y; a[6]=(_Float16)f1.z; a[7]=(_Float16)f1.w;
  a[8]=(_Float16)f2.x; a[9]=(_Float16)f2.y; a[10]=(_Float16)f2.z; a[11]=(_Float16)f2.w;
  a[12]=(_Float16)f3.x; a[13]=(_Float16)f3.y; a[14]=(_Float16)f3.z; a[15]=(_Float16)f3.w;
  return a;
}

__device__ __forceinline__ v8f wmma16(v16h a, v16h b, v8f c) {
  // 8 args: (neg_a, A, neg_b, B, c_mod, C, reuse_a, reuse_b)
  return __builtin_amdgcn_wmma_f32_16x16x32_f16(false, a, false, b, (short)0, c,
                                                false, false);
}

// -------------------------------------------------------------------------
// Kernel 1: value projection. v = value(:,b,:) (V x 256) @ W_val (256x256),
// output channel-innermost for the gather phase:
//   vimg[((b*NH + h)*V + p)*Dh + c]   with d = h*Dh + c.
// 8 waves/block, each wave = one 16x64 output tile (4 WMMA accumulators).
// -------------------------------------------------------------------------
__global__ void __launch_bounds__(256) value_proj_kernel(
    const float* __restrict__ value, const float* __restrict__ W_val,
    const float* __restrict__ b_val, float* __restrict__ vimg) {
  using namespace cfg;
  __shared__ _Float16 Wlds[64 * 256];  // [n][k], 32KB
  const int b = blockIdx.z;
  const int colbase = blockIdx.y * 64;
  const int t = threadIdx.x;
  {  // stage W panel: thread t loads row k=t, cols colbase..colbase+63
    const float* srcrow = W_val + (size_t)t * D + colbase;
#pragma unroll
    for (int j4 = 0; j4 < 64; j4 += 4) {
      float4 f = *(const float4*)(srcrow + j4);
      Wlds[(j4 + 0) * 256 + t] = (_Float16)f.x;
      Wlds[(j4 + 1) * 256 + t] = (_Float16)f.y;
      Wlds[(j4 + 2) * 256 + t] = (_Float16)f.z;
      Wlds[(j4 + 3) * 256 + t] = (_Float16)f.w;
    }
  }
  __syncthreads();
  const int wave = t >> 5, lane = t & 31;
  const int n = lane & 15, half = lane >> 4;
  const int kb = 16 * half, kbase = 16 * half, i0 = kbase >> 2;
  const int p0 = (blockIdx.x * 8 + wave) * 16;
  int prow = p0 + (lane & 15);
  if (prow >= V) prow = V - 1;  // clamp loads; stores guarded below
  const float* arow = value + ((size_t)prow * BS + b) * D;
  v8f acc[4] = {};
#pragma unroll
  for (int kk = 0; kk < 256; kk += 32) {
    const float4* ap = (const float4*)(arow + kk);
    v16h af = cvt_frag(ap[i0], ap[i0 + 1], ap[i0 + 2], ap[i0 + 3]);
    v16h bf[4];
#pragma unroll
    for (int u = 0; u < 4; ++u)
      bf[u] = *(const v16h*)&Wlds[(u * 16 + n) * 256 + kk + kb];
#pragma unroll
    for (int u = 0; u < 4; ++u) acc[u] = wmma16(af, bf[u], acc[u]);
  }
#pragma unroll
  for (int u = 0; u < 4; ++u) {
    const int d = colbase + u * 16 + n;
    const int h = d >> 5, c = d & (Dh - 1);
    const float bias = b_val[d];
    float* dst = vimg + ((size_t)(b * NH + h) * V) * Dh + c;
#pragma unroll
    for (int r = 0; r < 8; ++r) {
      int pr = p0 + r + 8 * half;
      if (pr < V) dst[(size_t)pr * Dh] = acc[u][r] + bias;
    }
  }
}

// -------------------------------------------------------------------------
// Kernel 2: q = query + query_pos, then off = q@W_off+b_off (3072x768) and
// attn logits = q@W_attn+b_attn (3072x384), fused by 16-wide column tile.
// -------------------------------------------------------------------------
__global__ void __launch_bounds__(256) q_proj_kernel(
    const float* __restrict__ query, const float* __restrict__ qpos,
    const float* __restrict__ W_off, const float* __restrict__ b_off,
    const float* __restrict__ W_attn, const float* __restrict__ b_attn,
    float* __restrict__ off_buf, float* __restrict__ aw_buf) {
  using namespace cfg;
  __shared__ _Float16 Wlds[16 * 256];
  const int t = threadIdx.x;
  const int tn = blockIdx.y;
  const float* Wp; const float* bp; float* dst; int ldw, colbase;
  if (tn < NOFF / 16) {
    Wp = W_off; bp = b_off; dst = off_buf; ldw = NOFF; colbase = tn * 16;
  } else {
    Wp = W_attn; bp = b_attn; dst = aw_buf; ldw = NATT;
    colbase = (tn - NOFF / 16) * 16;
  }
  {
    const float4* src = (const float4*)(Wp + (size_t)t * ldw + colbase);
    float4 f0 = src[0], f1 = src[1], f2 = src[2], f3 = src[3];
    float fv[16] = {f0.x,f0.y,f0.z,f0.w, f1.x,f1.y,f1.z,f1.w,
                    f2.x,f2.y,f2.z,f2.w, f3.x,f3.y,f3.z,f3.w};
#pragma unroll
    for (int j = 0; j < 16; ++j) Wlds[j * 256 + t] = (_Float16)fv[j];
  }
  __syncthreads();
  const int wave = t >> 5, lane = t & 31;
  const int n = lane & 15, half = lane >> 4;
  const int kb = 16 * half, kbase = 16 * half, i0 = kbase >> 2;
  const int row0 = (blockIdx.x * 8 + wave) * 16;
  const int row = row0 + (lane & 15);
  const float* q0 = query + (size_t)row * D;
  const float* q1 = qpos + (size_t)row * D;
  v8f acc = {};
#pragma unroll
  for (int kk = 0; kk < 256; kk += 32) {
    const float4* a0 = (const float4*)(q0 + kk);
    const float4* a1 = (const float4*)(q1 + kk);
    v16h af = cvt_frag(add4(a0[i0], a1[i0]), add4(a0[i0 + 1], a1[i0 + 1]),
                       add4(a0[i0 + 2], a1[i0 + 2]), add4(a0[i0 + 3], a1[i0 + 3]));
    v16h bf = *(const v16h*)&Wlds[n * 256 + kk + kb];
    acc = wmma16(af, bf, acc);
  }
  const float bias = bp[colbase + n];
#pragma unroll
  for (int r = 0; r < 8; ++r) {
    int rr = row0 + r + 8 * half;
    dst[(size_t)rr * ldw + colbase + n] = acc[r] + bias;
  }
}

// -------------------------------------------------------------------------
// Kernel 3: inline softmax over the 4 points + bilinear sampling of the BEV
// image (L2-resident) + weighted sum over points. One thread per (b, h, qq).
// Channel-innermost image layout -> each corner is 8x global_load_b128
// (128 contiguous bytes). Result written directly in the output-GEMM A
// layout: A_buf[r*3072 + s*256 + h*32 + c].
// -------------------------------------------------------------------------
__global__ void __launch_bounds__(256) sample_kernel(
    const float* __restrict__ off_buf, const float* __restrict__ aw_buf,
    const float* __restrict__ ref_trajs, const float* __restrict__ det_centers,
    const float* __restrict__ vimg, float* __restrict__ A_buf) {
  using namespace cfg;
  const int t = blockIdx.x * blockDim.x + threadIdx.x;  // 0 .. BS*NH*Q-1
  const int qq = t % Q;
  const int bh = t / Q;
  const int h = bh % NH, b = bh / NH;
  const int iq = qq / NS, s = qq % NS;
  const int a = iq / M, mm = iq % M;
  const int r = b * NQ + iq;

  const float* rt =
      ref_trajs + (((((size_t)b * A + a) * M + mm) * NS + (NS - 1)) * NL) * 2;
  const float cx = det_centers[((size_t)b * A + a) * 2 + 0];
  const float cy = det_centers[((size_t)b * A + a) * 2 + 1];
  const float rx = (rt[0] + cx + 51.2f) * (1.0f / 102.4f);
  const float ry = (rt[1] + cy + 51.2f) * (1.0f / 102.4f);

  // softmax over the NL*NP = 4 logits of this (h, s)
  const float* awp = aw_buf + (size_t)r * NATT + (h * NS + s) * NP;
  float lg[NP];
  float mx = -1e30f;
#pragma unroll
  for (int p = 0; p < NP; ++p) { lg[p] = awp[p]; mx = fmaxf(mx, lg[p]); }
  float sum = 0.f;
#pragma unroll
  for (int p = 0; p < NP; ++p) { lg[p] = __expf(lg[p] - mx); sum += lg[p]; }
  const float inv = 1.0f / sum;

  const float* ofp = off_buf + (size_t)r * NOFF + (size_t)(h * NS + s) * NP * 2;
  const float* img = vimg + ((size_t)(b * NH + h) * V) * Dh;  // [p][c]

  float acc[Dh];
#pragma unroll
  for (int c = 0; c < Dh; ++c) acc[c] = 0.f;

  for (int p = 0; p < NP; ++p) {
    const float wp = lg[p] * inv;
    const float lx = rx + ofp[p * 2 + 0] * (1.0f / WB);
    const float ly = ry + ofp[p * 2 + 1] * (1.0f / HB);
    const float x = lx * (float)WB - 0.5f;
    const float y = ly * (float)HB - 0.5f;
    const float x0f = floorf(x), y0f = floorf(y);
    const int x0 = (int)x0f, y0 = (int)y0f;
    const float wx1 = x - x0f, wx0 = 1.f - wx1;
    const float wy1 = y - y0f, wy0 = 1.f - wy1;
    const int xs[2] = {x0, x0 + 1}, ys[2] = {y0, y0 + 1};
    const float wxs[2] = {wx0, wx1}, wys[2] = {wy0, wy1};
#pragma unroll
    for (int j = 0; j < 2; ++j) {
#pragma unroll
      for (int i = 0; i < 2; ++i) {
        const int xi = xs[i], yi = ys[j];
        const bool valid = (xi >= 0) && (xi < WB) && (yi >= 0) && (yi < HB);
        const int xc = min(max(xi, 0), WB - 1);
        const int yc = min(max(yi, 0), HB - 1);
        const int idx = yc * WB + xc;
        const float wc = valid ? (wxs[i] * wys[j] * wp) : 0.f;
        const float* base = img + (size_t)idx * Dh;  // 128B contiguous
#pragma unroll
        for (int c4 = 0; c4 < Dh; c4 += 4) {
          float4 v4 = *(const float4*)(base + c4);
          acc[c4 + 0] += wc * v4.x;
          acc[c4 + 1] += wc * v4.y;
          acc[c4 + 2] += wc * v4.z;
          acc[c4 + 3] += wc * v4.w;
        }
      }
    }
  }

  float* dst = A_buf + (size_t)r * KOUT + s * D + h * Dh;
#pragma unroll
  for (int c = 0; c < Dh; c += 4) {
    float4 v4 = make_float4(acc[c], acc[c + 1], acc[c + 2], acc[c + 3]);
    *(float4*)(dst + c) = v4;
  }
}

// -------------------------------------------------------------------------
// Kernel 4: output projection o = A_buf (3072x3072) @ W_out (3072x256)+b_out.
// K chunked in 256-slices, 64-column LDS panel, 16x64 wave tile.
// -------------------------------------------------------------------------
__global__ void __launch_bounds__(256) out_proj_kernel(
    const float* __restrict__ A_buf, const float* __restrict__ W_out,
    const float* __restrict__ b_out, float* __restrict__ o_buf) {
  using namespace cfg;
  __shared__ _Float16 Wlds[64 * 256];  // 32KB
  const int t = threadIdx.x;
  const int colbase = blockIdx.y * 64;
  const int wave = t >> 5, lane = t & 31;
  const int n = lane & 15, half = lane >> 4;
  const int kb = 16 * half, kbase = 16 * half, i0 = kbase >> 2;
  const int row0 = (blockIdx.x * 8 + wave) * 16;
  const int row = row0 + (lane & 15);
  const float* arowbase = A_buf + (size_t)row * KOUT;
  v8f acc[4] = {};
  for (int kc = 0; kc < KOUT; kc += 256) {
    __syncthreads();
    {
      const float* srcrow = W_out + (size_t)(kc + t) * D + colbase;
#pragma unroll
      for (int j4 = 0; j4 < 64; j4 += 4) {
        float4 f = *(const float4*)(srcrow + j4);
        Wlds[(j4 + 0) * 256 + t] = (_Float16)f.x;
        Wlds[(j4 + 1) * 256 + t] = (_Float16)f.y;
        Wlds[(j4 + 2) * 256 + t] = (_Float16)f.z;
        Wlds[(j4 + 3) * 256 + t] = (_Float16)f.w;
      }
    }
    __syncthreads();
    const float* arow = arowbase + kc;
#pragma unroll
    for (int kk = 0; kk < 256; kk += 32) {
      const float4* ap = (const float4*)(arow + kk);
      v16h af = cvt_frag(ap[i0], ap[i0 + 1], ap[i0 + 2], ap[i0 + 3]);
      v16h bf[4];
#pragma unroll
      for (int u = 0; u < 4; ++u)
        bf[u] = *(const v16h*)&Wlds[(u * 16 + n) * 256 + kk + kb];
#pragma unroll
      for (int u = 0; u < 4; ++u) acc[u] = wmma16(af, bf[u], acc[u]);
    }
  }
#pragma unroll
  for (int u = 0; u < 4; ++u) {
    const int col = colbase + u * 16 + n;
    const float bias = b_out[col];
#pragma unroll
    for (int r = 0; r < 8; ++r) {
      int rr = row0 + r + 8 * half;
      o_buf[(size_t)rr * D + col] = acc[u][r] + bias;
    }
  }
}

// -------------------------------------------------------------------------
// Kernel 5: LayerNorm + ReLU + residual. One block per row of 256.
// -------------------------------------------------------------------------
__global__ void __launch_bounds__(256) ln_kernel(
    const float* __restrict__ o_buf, const float* __restrict__ query,
    const float* __restrict__ ln_g, const float* __restrict__ ln_b,
    float* __restrict__ out) {
  using namespace cfg;
  __shared__ float s1[256];
  __shared__ float s2[256];
  const int r = blockIdx.x, t = threadIdx.x;
  const float x = o_buf[(size_t)r * D + t];
  s1[t] = x;
  s2[t] = x * x;
  __syncthreads();
  for (int st = 128; st > 0; st >>= 1) {
    if (t < st) { s1[t] += s1[t + st]; s2[t] += s2[t + st]; }
    __syncthreads();
  }
  const float mu = s1[0] * (1.0f / D);
  const float var = s2[0] * (1.0f / D) - mu * mu;
  float y = (x - mu) * rsqrtf(var + 1e-5f) * ln_g[t] + ln_b[t];
  y = fmaxf(y, 0.f);
  out[(size_t)r * D + t] = y + query[(size_t)r * D + t];
}

extern "C" void kernel_launch(void* const* d_in, const int* in_sizes, int n_in,
                              void* d_out, int out_size, void* d_ws, size_t ws_size,
                              hipStream_t stream) {
  using namespace cfg;
  const float* query  = (const float*)d_in[0];
  const float* qpos   = (const float*)d_in[1];
  const float* value  = (const float*)d_in[2];
  const float* reft   = (const float*)d_in[3];
  const float* detc   = (const float*)d_in[4];
  const float* W_off  = (const float*)d_in[5];
  const float* b_off  = (const float*)d_in[6];
  const float* W_attn = (const float*)d_in[7];
  const float* b_attn = (const float*)d_in[8];
  const float* W_val  = (const float*)d_in[9];
  const float* b_val  = (const float*)d_in[10];
  const float* W_out  = (const float*)d_in[11];
  const float* b_out  = (const float*)d_in[12];
  const float* ln_g   = (const float*)d_in[13];
  const float* ln_b   = (const float*)d_in[14];
  float* out = (float*)d_out;

  // workspace layout (floats)
  float* vimg    = (float*)d_ws;                         // BS*NH*V*Dh  (81.9 MB)
  float* A_buf   = vimg + (size_t)BS * D * V;            // ROWS*KOUT   (37.7 MB)
  float* off_buf = A_buf + (size_t)ROWS * KOUT;          // ROWS*NOFF   ( 9.4 MB)
  float* aw_buf  = off_buf + (size_t)ROWS * NOFF;        // ROWS*NATT   ( 4.7 MB)
  float* o_buf   = aw_buf + (size_t)ROWS * NATT;         // ROWS*D      ( 3.1 MB)

  // 1) value projection -> channel-innermost BEV image
  value_proj_kernel<<<dim3(313, D / 64, BS), dim3(256), 0, stream>>>(
      value, W_val, b_val, vimg);
  // 2) q projections (off + attn fused by column tile)
  q_proj_kernel<<<dim3(ROWS / 128, (NOFF + NATT) / 16), dim3(256), 0, stream>>>(
      query, qpos, W_off, b_off, W_attn, b_attn, off_buf, aw_buf);
  // 3) softmax + bilinear sampling (one thread per (b,h,qq))
  sample_kernel<<<dim3((BS * NH * Q) / 256), dim3(256), 0, stream>>>(
      off_buf, aw_buf, reft, detc, vimg, A_buf);
  // 4) output projection
  out_proj_kernel<<<dim3(ROWS / 128, D / 64), dim3(256), 0, stream>>>(
      A_buf, W_out, b_out, o_buf);
  // 5) LayerNorm + ReLU + residual
  ln_kernel<<<dim3(ROWS), dim3(256), 0, stream>>>(o_buf, query, ln_g, ln_b, out);
}
